// Ernie4_5_MoeSparseMoeBlock_57801669870159
// MI455X (gfx1250) — compile-verified, matrix operand order checked by reading, not run on hardware
//
#include <hip/hip_runtime.h>
#include <hip/hip_bf16.h>

// ---------------- problem constants ----------------
#define H_DIM   1024
#define E_NUM   32
#define I_DIM   512
#define TWO_I   1024
#define TOPK    6
#define IS_DIM  1024
#define T_NUM   2048
#define PAIR_MAX (T_NUM * TOPK)      /* 12288 token-expert pairs */
#define PAIR_PAD (PAIR_MAX + 64)     /* slack for last partial tile */
#define LDA     136                  /* padded LDS row stride (bf16 elems) */

typedef __bf16 bf16;
typedef __attribute__((ext_vector_type(16))) __bf16 v16bf;
typedef __attribute__((ext_vector_type(8)))  float  v8f;
typedef unsigned int u32x4 __attribute__((ext_vector_type(4)));
typedef int          i32x4 __attribute__((ext_vector_type(4)));
typedef int          i32x8 __attribute__((ext_vector_type(8)));

// ---------------- WMMA helpers ----------------
__device__ __forceinline__ v8f wmma_bf16(v16bf a, v16bf b, v8f c) {
  // (neg_a, A, neg_b, B, c_mod, C, reuse_a, reuse_b)
  return __builtin_amdgcn_wmma_f32_16x16x32_bf16(false, a, false, b, (short)0, c,
                                                 false, false);
}

// A fragment (16x32 bf16, MxK) from LDS, row-major with stride ld.
// lane L: m = L&15 ; khi = (L>=16)?8:0 ; vgpr v -> k = (v>=4?16:0)+khi+(v&3)*2
__device__ __forceinline__ v16bf frag_a_lds(const bf16* xs, int ld, int row0,
                                            int k0, int lane) {
  const bf16* p = xs + (row0 + (lane & 15)) * ld + k0 + ((lane >> 4) << 3);
  v16bf a;
#pragma unroll
  for (int v = 0; v < 8; ++v) {
    int k = ((v & 4) << 2) + ((v & 3) << 1);
    a[2 * v]     = p[k];
    a[2 * v + 1] = p[k + 1];
  }
  return a;
}

// B fragment (32x16 bf16, KxN) where logical B[k][n] = W[n0+n][k0+k],
// W fp32 row-major with leading dim ld; convert fp32->bf16 in flight.
__device__ __forceinline__ v16bf frag_b_f32(const float* __restrict__ w, long ld,
                                            int n0, int k0, int lane) {
  const float* p = w + (long)(n0 + (lane & 15)) * ld + k0 + ((lane >> 4) << 3);
  __builtin_prefetch(p + 32, 0, 1);   // global_prefetch_b8 on the weight stream
  v16bf b;
#pragma unroll
  for (int v = 0; v < 8; ++v) {
    int k = ((v & 4) << 2) + ((v & 3) << 1);
    b[2 * v]     = (bf16)p[k];
    b[2 * v + 1] = (bf16)p[k + 1];
  }
  return b;
}

// Stage 64 x 128 fp32 -> bf16 into LDS (gathered rows via rtok, zero-fill pad).
__device__ __forceinline__ void stage_x(bf16* xs, const float* __restrict__ src,
                                        const int* rtok, int kc, int valid) {
  int tid = threadIdx.x;
  int cb = (tid & 31) << 2;           // 32 lanes * float4 = 128 cols
#pragma unroll
  for (int rr = 0; rr < 8; ++rr) {
    int row = (tid >> 5) + rr * 8;
    float4 v = make_float4(0.f, 0.f, 0.f, 0.f);
    if (row < valid)
      v = *(const float4*)(src + (long)rtok[row] * H_DIM + kc + cb);
    bf16* d = xs + row * LDA + cb;
    d[0] = (bf16)v.x; d[1] = (bf16)v.y; d[2] = (bf16)v.z; d[3] = (bf16)v.w;
  }
}

// Fallback stage 64 x 128 bf16 -> LDS (host pass / non-gfx1250).
__device__ __forceinline__ void stage_h(bf16* xs, const bf16* __restrict__ src,
                                        long srcld, int kc) {
  int tid = threadIdx.x;
  int cb = (tid & 31) << 2;           // 4 bf16 = 8 bytes
#pragma unroll
  for (int rr = 0; rr < 8; ++rr) {
    int row = (tid >> 5) + rr * 8;
    *(unsigned long long*)(xs + row * LDA + cb) =
        *(const unsigned long long*)(src + (long)row * srcld + kc + cb);
  }
}

#if defined(__gfx1250__)
// Tensor Data Mover: async DMA of a 64-row x 128-col bf16 tile (row stride
// srcld elems) from global into LDS at byte offset lds_off, inserting 4 DWORDs
// of padding after every 64 DWORDs (256B row) -> effective LDS stride = LDA.
__device__ __forceinline__ void tdm_load_tile_bf16(const bf16* gsrc,
                                                   long srcld_elems,
                                                   unsigned lds_off) {
  unsigned long long ga = (unsigned long long)(size_t)gsrc;
  u32x4 g0;
  g0.x = 1u;                                     // count=1, user descriptor
  g0.y = lds_off;                                // lds_addr [63:32]
  g0.z = (unsigned)(ga & 0xffffffffu);           // global_addr lo
  g0.w = (unsigned)((ga >> 32) & 0x01ffffffu)    // global_addr hi (57-bit)
       | (2u << 30);                             // type = 2 ("image")
  i32x8 g1;
  g1[0] = (1 << 16)        /* data_size = 2B */
        | (1 << 20)        /* pad_enable  */
        | (5 << 22)        /* pad_interval: 64 DWORDs */
        | (3 << 25);       /* pad_amount : 4 DWORDs  */
  g1[1] = (int)((srcld_elems & 0xffffu) << 16);  // tensor_dim0 lo16 @ [63:48]
  g1[2] = (int)((srcld_elems >> 16) & 0xffffu)   // tensor_dim0 hi16
        | (64 << 16);                            // tensor_dim1 lo16 = 64 rows
  g1[3] = (128 << 16);                           // tile_dim0 = 128 @ [127:112]
  g1[4] = 64;                                    // tile_dim1 = 64
  g1[5] = (int)srcld_elems;                      // tensor_dim0_stride lo32
  g1[6] = 0;
  g1[7] = 0;
  i32x4 gz = {0, 0, 0, 0};
#if __clang_major__ >= 23
  i32x8 gz8 = {0, 0, 0, 0, 0, 0, 0, 0};
  __builtin_amdgcn_tensor_load_to_lds(g0, g1, gz, gz, gz8, 0);
#else
  __builtin_amdgcn_tensor_load_to_lds(g0, g1, gz, gz, 0);
#endif
}
#endif

__device__ __forceinline__ float silu(float g) {
  return g / (1.f + __expf(-g));
}

// ---------------- router: softmax + top-6 ----------------
__global__ __launch_bounds__(256) void router_kernel(
    const float* __restrict__ x, const float* __restrict__ rw,
    const float* __restrict__ bias, int* __restrict__ topk_idx,
    float* __restrict__ topk_w) {
  int lane = threadIdx.x & 31, wv = threadIdx.x >> 5;
  int t = blockIdx.x * 8 + wv;
  const float4* xr = (const float4*)(x + (long)t * H_DIM);
  const float4* wr = (const float4*)(rw + (long)lane * H_DIM);
  float acc = 0.f;
  for (int i = 0; i < H_DIM / 4; ++i) {
    float4 a = xr[i], b = wr[i];
    acc += a.x * b.x + a.y * b.y + a.z * b.z + a.w * b.w;
  }
  float mx = acc;
  for (int off = 16; off > 0; off >>= 1) mx = fmaxf(mx, __shfl_xor(mx, off, 32));
  float p = __expf(acc - mx);
  float s = p;
  for (int off = 16; off > 0; off >>= 1) s += __shfl_xor(s, off, 32);
  float prob = p / s;
  float c = prob + bias[lane];
  float wacc = 0.f;
  int   myidx[TOPK];
  float myw[TOPK];
  for (int j = 0; j < TOPK; ++j) {
    float v = c; int id = lane;
    for (int off = 16; off > 0; off >>= 1) {
      float ov = __shfl_xor(v, off, 32);
      int   oi = __shfl_xor(id, off, 32);
      if (ov > v || (ov == v && oi < id)) { v = ov; id = oi; }
    }
    float pw = __shfl(prob, id, 32);
    myidx[j] = id; myw[j] = pw; wacc += pw;
    if (lane == id) c = -3.4e38f;
  }
  wacc = fmaxf(wacc, 1e-12f);
  if (lane < TOPK) {
    topk_idx[t * TOPK + lane] = myidx[lane];
    topk_w[t * TOPK + lane]   = myw[lane] / wacc;
  }
}

// ---------------- bucketing ----------------
__global__ void zero32_kernel(int* counts, int* cursor) {
  if (threadIdx.x < E_NUM) { counts[threadIdx.x] = 0; cursor[threadIdx.x] = 0; }
}
__global__ void count_kernel(const int* __restrict__ topk_idx, int* counts) {
  int i = blockIdx.x * 256 + threadIdx.x;
  if (i < PAIR_MAX) atomicAdd(&counts[topk_idx[i]], 1);
}
__global__ void scan_kernel(const int* __restrict__ counts, int* ebase) {
  if (threadIdx.x == 0) {
    int s = 0;
    for (int e = 0; e < E_NUM; ++e) { ebase[e] = s; s += counts[e]; }
  }
}
__global__ void scatter_kernel(const int* __restrict__ topk_idx,
                               const float* __restrict__ topk_w,
                               const int* __restrict__ ebase, int* cursor,
                               int* pair_tok, float* pair_w) {
  int i = blockIdx.x * 256 + threadIdx.x;
  if (i < PAIR_MAX) {
    int e = topk_idx[i];
    int pos = ebase[e] + atomicAdd(&cursor[e], 1);
    pair_tok[pos] = i / TOPK;
    pair_w[pos]   = topk_w[i];
  }
}

// ---------------- MoE gate_up + SiLU: hmid[pair, I] ----------------
// grid (E, I/128), block 256. Each wave owns a 16-wide column slice and
// computes matching gate and up columns so SiLU fuses in-register.
__global__ __launch_bounds__(256) void moe_gateup_kernel(
    const float* __restrict__ x, const float* __restrict__ gup,
    const int* __restrict__ pair_tok, const int* __restrict__ ebase,
    const int* __restrict__ ecnt, bf16* __restrict__ hmid) {
  int e = blockIdx.x;
  int lane = threadIdx.x & 31, wv = threadIdx.x >> 5;
  int nc = blockIdx.y * 128 + wv * 16;
  const float* Wg = gup + (long)e * TWO_I * H_DIM;
  const float* Wu = Wg + (long)I_DIM * H_DIM;
  int cnt = ecnt[e], base = ebase[e];
  int ntile = (cnt + 63) >> 6;
  __shared__ bf16 xs[64 * LDA];
  __shared__ int  rtok[64];
  int n = lane & 15, hi8 = (lane >> 4) << 3;
  for (int t = 0; t < ntile; ++t) {
    int row0 = base + t * 64;
    int valid = min(64, cnt - t * 64);
    if (threadIdx.x < 64)
      rtok[threadIdx.x] = (threadIdx.x < valid) ? pair_tok[row0 + threadIdx.x] : 0;
    v8f accg[4], accu[4];
    v8f z = {0.f, 0.f, 0.f, 0.f, 0.f, 0.f, 0.f, 0.f};
#pragma unroll
    for (int m = 0; m < 4; ++m) { accg[m] = z; accu[m] = z; }
    for (int kc = 0; kc < H_DIM; kc += 128) {
      __syncthreads();
      stage_x(xs, x, rtok, kc, valid);
      __syncthreads();
#pragma unroll
      for (int kk = 0; kk < 128; kk += 32) {
        v16bf bg = frag_b_f32(Wg, H_DIM, nc, kc + kk, lane);
        v16bf bu = frag_b_f32(Wu, H_DIM, nc, kc + kk, lane);
#pragma unroll
        for (int m = 0; m < 4; ++m) {
          v16bf a = frag_a_lds(xs, LDA, m * 16, kk, lane);
          accg[m] = wmma_bf16(a, bg, accg[m]);
          accu[m] = wmma_bf16(a, bu, accu[m]);
        }
      }
    }
#pragma unroll
    for (int m = 0; m < 4; ++m)
#pragma unroll
      for (int r = 0; r < 8; ++r) {
        int row = m * 16 + hi8 + r;
        if (row < valid) {
          float hv = silu(accg[m][r]) * accu[m][r];
          hmid[(long)(row0 + row) * I_DIM + nc + n] = (bf16)hv;
        }
      }
    __syncthreads();
  }
}

// ---------------- MoE down proj + weighted scatter-add ----------------
// grid (E, H/128), block 256. A tiles are contiguous bf16 rows -> staged by
// the Tensor Data Mover with double-buffered LDS; s_wait_tensorcnt 1 after
// issuing tile i+1 guarantees tile i is resident (TDM ops complete in order).
__global__ __launch_bounds__(256) void moe_down_kernel(
    const bf16* __restrict__ hmid, const float* __restrict__ dproj,
    const int* __restrict__ pair_tok, const float* __restrict__ pair_w,
    const int* __restrict__ ebase, const int* __restrict__ ecnt,
    float* __restrict__ out) {
  int e = blockIdx.x;
  int lane = threadIdx.x & 31, wv = threadIdx.x >> 5;
  int nc = blockIdx.y * 128 + wv * 16;
  const float* Wd = dproj + (long)e * H_DIM * I_DIM;
  int cnt = ecnt[e], base = ebase[e];
  int ntile = (cnt + 63) >> 6;
  __shared__ bf16  xs[2][64 * LDA];
  __shared__ int   rtok[64];
  __shared__ float rwt[64];
  int n = lane & 15, hi8 = (lane >> 4) << 3;
  for (int t = 0; t < ntile; ++t) {
    int row0 = base + t * 64;
    int valid = min(64, cnt - t * 64);
    if (threadIdx.x < 64) {
      bool ok = threadIdx.x < valid;
      rtok[threadIdx.x] = ok ? pair_tok[row0 + threadIdx.x] : 0;
      rwt[threadIdx.x]  = ok ? pair_w[row0 + threadIdx.x] : 0.f;
    }
    v8f acc[4];
    v8f z = {0.f, 0.f, 0.f, 0.f, 0.f, 0.f, 0.f, 0.f};
#pragma unroll
    for (int m = 0; m < 4; ++m) acc[m] = z;
    const bf16* src = hmid + (long)row0 * I_DIM;
#if defined(__gfx1250__)
    if (wv == 0) tdm_load_tile_bf16(src, I_DIM, (unsigned)(size_t)&xs[0][0]);
#endif
    for (int i = 0; i < I_DIM / 128; ++i) {
      int buf = i & 1;
#if defined(__gfx1250__)
      if (wv == 0) {
        if (i + 1 < I_DIM / 128) {
          tdm_load_tile_bf16(src + (i + 1) * 128, I_DIM,
                             (unsigned)(size_t)&xs[buf ^ 1][0]);
          __builtin_amdgcn_s_wait_tensorcnt(1);
        } else {
          __builtin_amdgcn_s_wait_tensorcnt(0);
        }
      }
#else
      stage_h(xs[buf], src, I_DIM, i * 128);
#endif
      __syncthreads();
#pragma unroll
      for (int kk = 0; kk < 128; kk += 32) {
        v16bf b = frag_b_f32(Wd, I_DIM, nc, i * 128 + kk, lane);
#pragma unroll
        for (int m = 0; m < 4; ++m) {
          v16bf a = frag_a_lds(xs[buf], LDA, m * 16, kk, lane);
          acc[m] = wmma_bf16(a, b, acc[m]);
        }
      }
      __syncthreads();
    }
#pragma unroll
    for (int m = 0; m < 4; ++m)
#pragma unroll
      for (int r = 0; r < 8; ++r) {
        int row = m * 16 + hi8 + r;
        if (row < valid)
          atomicAdd(out + (long)rtok[row] * H_DIM + nc + n, rwt[row] * acc[m][r]);
      }
    __syncthreads();
  }
}

// ---------------- shared expert gate/up: smid[T, IS] ----------------
// grid (T/64, IS/128), block 256.
__global__ __launch_bounds__(256) void shared_gateup_kernel(
    const float* __restrict__ x, const float* __restrict__ wg,
    const float* __restrict__ wu, bf16* __restrict__ smid) {
  int t0 = blockIdx.x * 64;
  int lane = threadIdx.x & 31, wv = threadIdx.x >> 5;
  int nc = blockIdx.y * 128 + wv * 16;
  __shared__ bf16 xs[64 * LDA];
  __shared__ int  rtok[64];
  if (threadIdx.x < 64) rtok[threadIdx.x] = t0 + threadIdx.x;
  int n = lane & 15, hi8 = (lane >> 4) << 3;
  v8f accg[4], accu[4];
  v8f z = {0.f, 0.f, 0.f, 0.f, 0.f, 0.f, 0.f, 0.f};
#pragma unroll
  for (int m = 0; m < 4; ++m) { accg[m] = z; accu[m] = z; }
  for (int kc = 0; kc < H_DIM; kc += 128) {
    __syncthreads();
    stage_x(xs, x, rtok, kc, 64);
    __syncthreads();
#pragma unroll
    for (int kk = 0; kk < 128; kk += 32) {
      v16bf bg = frag_b_f32(wg, H_DIM, nc, kc + kk, lane);
      v16bf bu = frag_b_f32(wu, H_DIM, nc, kc + kk, lane);
#pragma unroll
      for (int m = 0; m < 4; ++m) {
        v16bf a = frag_a_lds(xs, LDA, m * 16, kk, lane);
        accg[m] = wmma_bf16(a, bg, accg[m]);
        accu[m] = wmma_bf16(a, bu, accu[m]);
      }
    }
  }
#pragma unroll
  for (int m = 0; m < 4; ++m)
#pragma unroll
    for (int r = 0; r < 8; ++r) {
      int row = m * 16 + hi8 + r;
      float hv = silu(accg[m][r]) * accu[m][r];
      smid[(long)(t0 + row) * IS_DIM + nc + n] = (bf16)hv;
    }
}

// ---------------- shared expert down: out[T, H] = smid @ Wd^T (writes) ----
// grid (T/64, H/128), block 256. TDM double-buffered staging as above.
__global__ __launch_bounds__(256) void shared_down_kernel(
    const bf16* __restrict__ smid, const float* __restrict__ wd,
    float* __restrict__ out) {
  int t0 = blockIdx.x * 64;
  int lane = threadIdx.x & 31, wv = threadIdx.x >> 5;
  int nc = blockIdx.y * 128 + wv * 16;
  __shared__ bf16 xs[2][64 * LDA];
  int n = lane & 15, hi8 = (lane >> 4) << 3;
  v8f acc[4];
  v8f z = {0.f, 0.f, 0.f, 0.f, 0.f, 0.f, 0.f, 0.f};
#pragma unroll
  for (int m = 0; m < 4; ++m) acc[m] = z;
  const bf16* src = smid + (long)t0 * IS_DIM;
#if defined(__gfx1250__)
  if (wv == 0) tdm_load_tile_bf16(src, IS_DIM, (unsigned)(size_t)&xs[0][0]);
#endif
  for (int i = 0; i < IS_DIM / 128; ++i) {
    int buf = i & 1;
#if defined(__gfx1250__)
    if (wv == 0) {
      if (i + 1 < IS_DIM / 128) {
        tdm_load_tile_bf16(src + (i + 1) * 128, IS_DIM,
                           (unsigned)(size_t)&xs[buf ^ 1][0]);
        __builtin_amdgcn_s_wait_tensorcnt(1);
      } else {
        __builtin_amdgcn_s_wait_tensorcnt(0);
      }
    }
#else
    stage_h(xs[buf], src, IS_DIM, i * 128);
#endif
    __syncthreads();
#pragma unroll
    for (int kk = 0; kk < 128; kk += 32) {
      v16bf b = frag_b_f32(wd, IS_DIM, nc, i * 128 + kk, lane);
#pragma unroll
      for (int m = 0; m < 4; ++m) {
        v16bf a = frag_a_lds(xs[buf], LDA, m * 16, kk, lane);
        acc[m] = wmma_bf16(a, b, acc[m]);
      }
    }
    __syncthreads();
  }
#pragma unroll
  for (int m = 0; m < 4; ++m)
#pragma unroll
    for (int r = 0; r < 8; ++r) {
      int row = m * 16 + hi8 + r;
      out[(long)(t0 + row) * H_DIM + nc + n] = acc[m][r];
    }
}

// ---------------- host launcher ----------------
static inline size_t align256(size_t x) { return (x + 255) & ~(size_t)255; }

extern "C" void kernel_launch(void* const* d_in, const int* in_sizes, int n_in,
                              void* d_out, int out_size, void* d_ws, size_t ws_size,
                              hipStream_t stream) {
  const float* x     = (const float*)d_in[0];  // [B,S,H]
  const float* rw    = (const float*)d_in[1];  // [E,H]
  const float* ebias = (const float*)d_in[2];  // [E]
  const float* gup   = (const float*)d_in[3];  // [E,2I,H]
  const float* dpr   = (const float*)d_in[4];  // [E,H,I]
  const float* swg   = (const float*)d_in[5];  // [IS,H]
  const float* swu   = (const float*)d_in[6];  // [IS,H]
  const float* swd   = (const float*)d_in[7];  // [H,IS]
  float* out = (float*)d_out;

  // workspace layout
  char* ws = (char*)d_ws;
  size_t off = 0;
  int*   topk_idx = (int*)(ws + off);   off = align256(off + PAIR_MAX * 4);
  float* topk_w   = (float*)(ws + off); off = align256(off + PAIR_MAX * 4);
  int*   counts   = (int*)(ws + off);   off = align256(off + E_NUM * 4);
  int*   ebase    = (int*)(ws + off);   off = align256(off + E_NUM * 4);
  int*   cursor   = (int*)(ws + off);   off = align256(off + E_NUM * 4);
  int*   pair_tok = (int*)(ws + off);   off = align256(off + PAIR_PAD * 4);
  float* pair_w   = (float*)(ws + off); off = align256(off + PAIR_PAD * 4);
  bf16*  smid     = (bf16*)(ws + off);  off = align256(off + (size_t)T_NUM * IS_DIM * 2);
  bf16*  hmid     = (bf16*)(ws + off);  off = align256(off + (size_t)PAIR_PAD * I_DIM * 2);
  (void)ws_size; (void)in_sizes; (void)n_in; (void)out_size;

  // 1. routing
  router_kernel<<<T_NUM / 8, 256, 0, stream>>>(x, rw, ebias, topk_idx, topk_w);
  // 2. bucket tokens by expert
  zero32_kernel<<<1, 32, 0, stream>>>(counts, cursor);
  count_kernel<<<PAIR_MAX / 256, 256, 0, stream>>>(topk_idx, counts);
  scan_kernel<<<1, 32, 0, stream>>>(counts, ebase);
  scatter_kernel<<<PAIR_MAX / 256, 256, 0, stream>>>(topk_idx, topk_w, ebase,
                                                     cursor, pair_tok, pair_w);
  // 3. shared experts (down kernel WRITES out, establishing the base)
  shared_gateup_kernel<<<dim3(T_NUM / 64, IS_DIM / 128), 256, 0, stream>>>(
      x, swg, swu, smid);
  shared_down_kernel<<<dim3(T_NUM / 64, H_DIM / 128), 256, 0, stream>>>(
      smid, swd, out);
  // 4. sparse expert MLP, accumulates into out
  moe_gateup_kernel<<<dim3(E_NUM, I_DIM / 128), 256, 0, stream>>>(
      x, gup, pair_tok, ebase, counts, hmid);
  moe_down_kernel<<<dim3(E_NUM, H_DIM / 128), 256, 0, stream>>>(
      hmid, dpr, pair_tok, pair_w, ebase, counts, out);
}
